// MSDBlock2d_62766652063752
// MI455X (gfx1250) — matrix operand order repaired and checked by reading, not compiled, not run on hardware
//
#include <hip/hip_runtime.h>
#include <stdint.h>

#define HH 384
#define WW 384
#define BATCH 4
#define INCH 32
#define GROW 16
#define DEPTH_L 12
#define CTOT 224                       // 32 + 12*16
#define MIRROR_ELEMS ((size_t)BATCH * HH * WW * CTOT)   // 132,120,576 bf16 elems

typedef __attribute__((ext_vector_type(16))) __bf16 bf16x16;
typedef __attribute__((ext_vector_type(8)))  float  v8f;
typedef __attribute__((ext_vector_type(8)))  unsigned short u16x8;

__device__ __forceinline__ unsigned short f2bf(float f) {
    union { float f; uint32_t u; } v; v.f = f;
    uint32_t u = v.u;
    uint32_t r = (u + 0x7FFFu + ((u >> 16) & 1u)) >> 16;   // RNE
    return (unsigned short)r;
}

// Copy f32 input NCHW[4,32,384,384] -> d_out NCHW[4,224,..] channels 0..31,
// and write bf16 NHWC mirror.
__global__ void pack_input_kernel(const float* __restrict__ in,
                                  float* __restrict__ out,
                                  unsigned short* __restrict__ mir) {
    int idx = blockIdx.x * 256 + threadIdx.x;     // exactly B*32*H*W threads
    int x = idx % WW; int t = idx / WW;
    int y = t % HH;   t /= HH;
    int c = t % INCH; int b = t / INCH;
    float v = in[idx];
    out[((b * CTOT + c) * HH + y) * WW + x] = v;
    mir[(((size_t)(b * HH + y) * WW + x) * CTOT) + c] = f2bf(v);
}

// Pack f32 weights [16, in_ch, 3, 3] into bf16 A-fragments, layout
// [tap][chunk][lane][elem] so each lane reads 32 contiguous bytes.
// A 16x32 layout (contiguous k, matching B loads): elem e ->
// K = 16*(lane/16) + e ; M = lane%16.
__global__ void pack_weights_kernel(const float* __restrict__ w,
                                    unsigned short* __restrict__ dst,
                                    int in_ch, int nChunks) {
    int t = blockIdx.x * 256 + threadIdx.x;
    int total = 9 * nChunks * 512;
    if (t >= total) return;
    int tap   = t / (nChunks * 512);
    int r     = t % (nChunks * 512);
    int chunk = r >> 9;
    int le    = r & 511;
    int lane  = le >> 4;
    int e     = le & 15;
    int m     = lane & 15;
    int grp   = lane >> 4;
    int j = e >> 1, hh = e & 1;
    (void)j; (void)hh;
    int k  = 16 * grp + e;
    int c  = chunk * 32 + k;
    float v = (c < in_ch) ? w[(m * in_ch + c) * 9 + tap] : 0.0f;
    dst[t] = f2bf(v);
}

__device__ __forceinline__ int reflect(int t, int n) {
    t = t < 0 ? -t : t;
    return t >= n ? 2 * (n - 1) - t : t;
}

// One dilated-conv layer as implicit GEMM.
// Each wave: 16 outch x 32 pixels = two N-tiles sharing one A-fragment,
// K = 9 taps x ceil(in_ch/32) chunks of bf16 WMMA (f32 accumulate).
// Channel overhang (in_ch % 32 == 16) handled by a wave-uniform tail chunk.
__global__ __launch_bounds__(128)
void msd_layer_kernel(const unsigned short* __restrict__ mir,   // NHWC bf16 (read+write)
                      unsigned short* __restrict__ mir_out,
                      const unsigned short* __restrict__ wpack, // layer's packed weights
                      const float* __restrict__ bias,           // layer's 16 biases
                      float* __restrict__ out,
                      int in_ch, int nChunks, int dil, int layer) {
    const int lane = threadIdx.x & 31;
    const int wv   = threadIdx.x >> 5;
    const int x0   = blockIdx.x * 128 + wv * 32;
    const int y    = blockIdx.y;
    const int b    = blockIdx.z;
    const int n    = lane & 15;
    const int grp  = lane >> 4;

    const int nFull    = in_ch >> 5;                 // full 32-channel chunks
    const bool hasTail = (in_ch & 31) != 0;          // wave-uniform
    const int cbTail   = grp ? 0 : (nFull << 5);     // overhang -> finite data, A=0 there
    const int cbFull   = grp << 4;                   // +16 channels for upper lane half

    v8f acc0 = {};
    v8f acc1 = {};
    for (int ky = 0; ky < 3; ++ky) {
        const int yy = reflect(y + (ky - 1) * dil, HH);
        const size_t rowbase = (size_t)(b * HH + yy) * WW;
        for (int kx = 0; kx < 3; ++kx) {
            const int xs = (kx - 1) * dil;
            const int xx0 = reflect(x0 + n + xs, WW);
            const int xx1 = reflect(x0 + 16 + n + xs, WW);
            const unsigned short* p0 = mir + (rowbase + xx0) * CTOT + cbFull;
            const unsigned short* p1 = mir + (rowbase + xx1) * CTOT + cbFull;
            const int tap = ky * 3 + kx;
            const unsigned short* wtap =
                wpack + ((size_t)(tap * nChunks) << 9) + (lane << 4);
            for (int chunk = 0; chunk < nFull; ++chunk) {
                bf16x16 b0 = *(const bf16x16*)(p0 + (chunk << 5));
                bf16x16 b1 = *(const bf16x16*)(p1 + (chunk << 5));
                bf16x16 a  = *(const bf16x16*)(wtap + ((size_t)chunk << 9));
                acc0 = __builtin_amdgcn_wmma_f32_16x16x32_bf16(
                    false, a, false, b0, (short)0, acc0, false, false);
                acc1 = __builtin_amdgcn_wmma_f32_16x16x32_bf16(
                    false, a, false, b1, (short)0, acc1, false, false);
            }
            if (hasTail) {                           // scalar branch; EXEC unchanged
                bf16x16 b0 = *(const bf16x16*)(mir + (rowbase + xx0) * CTOT + cbTail);
                bf16x16 b1 = *(const bf16x16*)(mir + (rowbase + xx1) * CTOT + cbTail);
                bf16x16 a  = *(const bf16x16*)(wtap + ((size_t)nFull << 9));
                acc0 = __builtin_amdgcn_wmma_f32_16x16x32_bf16(
                    false, a, false, b0, (short)0, acc0, false, false);
                acc1 = __builtin_amdgcn_wmma_f32_16x16x32_bf16(
                    false, a, false, b1, (short)0, acc1, false, false);
            }
        }
    }

    // Epilogue: D layout -> lane = pixel n, vgpr v -> M = v + 8*grp.
    const int mbase   = 8 * grp;
    const int ochbase = INCH + GROW * layer;
    const int xA = x0 + n;
    const int xB = x0 + 16 + n;
    u16x8 pkA, pkB;
    #pragma unroll
    for (int v = 0; v < 8; ++v) {
        const int m = mbase + v;
        const float bm = bias[m];
        float vA = acc0[v] + bm; vA = vA > 0.0f ? vA : 0.0f;
        float vB = acc1[v] + bm; vB = vB > 0.0f ? vB : 0.0f;
        const size_t chb = (size_t)(b * CTOT + ochbase + m) * HH + y;
        out[chb * WW + xA] = vA;
        out[chb * WW + xB] = vB;
        pkA[v] = f2bf(vA);
        pkB[v] = f2bf(vB);
    }
    const size_t rowpix = (size_t)(b * HH + y) * WW;
    *(u16x8*)(mir_out + (rowpix + xA) * CTOT + ochbase + mbase) = pkA;
    *(u16x8*)(mir_out + (rowpix + xB) * CTOT + ochbase + mbase) = pkB;
}

extern "C" void kernel_launch(void* const* d_in, const int* in_sizes, int n_in,
                              void* d_out, int out_size, void* d_ws, size_t ws_size,
                              hipStream_t stream) {
    (void)in_sizes; (void)n_in; (void)out_size; (void)ws_size;
    const float* input = (const float*)d_in[0];
    const float* bias  = (const float*)d_in[1];
    float* out = (float*)d_out;

    unsigned short* mir   = (unsigned short*)d_ws;          // 264 MB bf16 NHWC mirror
    unsigned short* wpack = mir + MIRROR_ELEMS;             // packed bf16 weights after it

    // 1) Stage input into d_out (NCHW, C=224) and bf16 NHWC mirror.
    const int nin = BATCH * INCH * HH * WW;                 // multiple of 256
    pack_input_kernel<<<nin / 256, 256, 0, stream>>>(input, out, mir);

    // 2) Pack all 12 layers' weights into WMMA A-fragment order.
    static const int dil[DEPTH_L] = {1, 2, 3, 4, 5, 6, 7, 8, 9, 10, 1, 2};
    size_t bases[DEPTH_L]; int chunks[DEPTH_L];
    size_t wbase = 0;
    for (int i = 0; i < DEPTH_L; ++i) {
        const int in_ch = INCH + GROW * i;
        const int nC = (in_ch + 31) / 32;
        bases[i] = wbase; chunks[i] = nC;
        const int total = 9 * nC * 512;
        pack_weights_kernel<<<(total + 255) / 256, 256, 0, stream>>>(
            (const float*)d_in[2 + i], wpack + wbase, in_ch, nC);
        wbase += (size_t)total;
    }

    // 3) 12 dependent implicit-GEMM conv layers.
    dim3 grid(WW / 128, HH, BATCH);   // 4 waves/block, 32 px per wave
    for (int i = 0; i < DEPTH_L; ++i) {
        const int in_ch = INCH + GROW * i;
        msd_layer_kernel<<<grid, 128, 0, stream>>>(
            mir, mir, wpack + bases[i], bias + 16 * i, out,
            in_ch, chunks[i], dil[i], i);
    }
}
